// MLPSoftmaxRouter_39745627357229
// MI455X (gfx1250) — matrix operand order repaired and checked by reading, not compiled, not run on hardware
//
#include <hip/hip_runtime.h>
#include <hip/hip_bf16.h>

typedef __attribute__((ext_vector_type(16))) __bf16 v16bf;
typedef __attribute__((ext_vector_type(8)))  __bf16 v8bf;
typedef __attribute__((ext_vector_type(2)))  __bf16 v2bf;
typedef __attribute__((ext_vector_type(8)))  float  v8f;
typedef __attribute__((ext_vector_type(4)))  float  f32x4;
typedef __attribute__((ext_vector_type(4)))  unsigned int u32x4;
typedef __attribute__((ext_vector_type(8)))  int i32x8;
typedef __attribute__((ext_vector_type(4)))  int i32x4;

// ---- LDS layout (bytes) ----
#define XS_OFF   0          // x tile bf16 [128][1032]
#define XS_STR   1032
#define W1_OFF   264192     // W1 chunk bf16 [64][264]  (K=256 staged)
#define W1_STR   264
#define W2_OFF   297984     // W2 chunk bf16 [64][72]
#define W2_STR   72
#define HS_OFF   307200     // per-wave h tile bf16 [16][72]
#define HS_WV    2304
#define HS_STR   144        // bytes per row
#define LG_OFF   0          // logits f32 [128][68] (overlays XS)
#define LG_STR   68
#define RI_OFF   36864      // per-row {p1,p2,i1,i2}
#define SMEM_BYTES 325632

#if __has_builtin(__builtin_amdgcn_tensor_load_to_lds) && \
    __has_builtin(__builtin_amdgcn_s_wait_tensorcnt)
#define HAS_TDM 1
#else
#define HAS_TDM 0
#endif

// fp32 -> packed 2x bf16. Let the backend use a native cvt if gfx1250 has one;
// otherwise clang emits the RNE software sequence (equivalent to manual trick).
static __device__ __forceinline__ unsigned pk2bf(float a, float b) {
  union { v2bf v; unsigned u; } c;
  c.v = (v2bf){(__bf16)a, (__bf16)b};
  return c.u;
}
static __device__ __forceinline__ unsigned short f2bf(float f) {
  union { __bf16 h; unsigned short s; } c;
  c.h = (__bf16)f;
  return c.s;
}

union QV { uint4 q; v8bf v; };
static __device__ __forceinline__ v8bf ldsv8(const char* p) {
  QV u; u.q = *(const uint4*)p; return u.v;
}
static __device__ __forceinline__ v16bf mk16(v8bf lo, v8bf hi) {
  return __builtin_shufflevector(lo, hi, 0,1,2,3,4,5,6,7,8,9,10,11,12,13,14,15);
}
static __device__ __forceinline__ float gelu_exact(float v) {
  return 0.5f * v * (1.0f + erff(v * 0.70710678118654752f));
}

#if HAS_TDM
// Issue one 2D TDM tile load (bf16 data, padded LDS rows) per the CDNA5 D# layout.
// 6-arg builtin form (amdgpu-toolchain / clang-23).
static __device__ __forceinline__ void tdm_load_2d_bf16(
    unsigned lds_addr, const void* gaddr,
    unsigned tensor_d0, unsigned tensor_d1, unsigned long stride0_elems,
    unsigned tile_d0, unsigned tile_d1,
    unsigned pad_interval_code, unsigned pad_amount_code)
{
  unsigned long ga = (unsigned long)(uintptr_t)gaddr;
  u32x4 g0;
  g0.x = 1u;                                         // count=1, user mode
  g0.y = lds_addr;                                   // LDS byte address
  g0.z = (unsigned)(ga & 0xFFFFFFFFu);               // global_addr[31:0]
  g0.w = (unsigned)((ga >> 32) & 0x1FFFFFFu) | (2u << 30); // addr[56:32] | type=2
  i32x8 g1;
  g1[0] = (int)((1u << 16) | (1u << 20) |            // data_size=2B, pad_enable
                (pad_interval_code << 22) | (pad_amount_code << 25));
  g1[1] = (int)((tensor_d0 & 0xFFFFu) << 16);        // tensor_dim0[15:0]
  g1[2] = (int)((tensor_d0 >> 16) | ((tensor_d1 & 0xFFFFu) << 16));
  g1[3] = (int)((tensor_d1 >> 16) | (tile_d0 << 16));
  g1[4] = (int)tile_d1;                              // tile_dim2 = 0
  g1[5] = (int)(stride0_elems & 0xFFFFFFFFu);
  g1[6] = (int)((stride0_elems >> 32) & 0xFFFFu);    // dim1_stride unused (2D)
  g1[7] = 0;
  i32x4 z4 = {0, 0, 0, 0};
  i32x8 z8 = {0, 0, 0, 0, 0, 0, 0, 0};
  __builtin_amdgcn_tensor_load_to_lds(g0, g1, z4, z4, z8, 0);
}
#endif

// ---- pre-pass: fp32 -> bf16 weight conversion (once per launch) ----
__global__ __launch_bounds__(256) void cvt_w_bf16(const float4* __restrict__ src,
                                                  uint2* __restrict__ dst, int n4) {
  int i = blockIdx.x * blockDim.x + threadIdx.x;
  if (i < n4) {
    float4 v = src[i];
    dst[i] = make_uint2(pk2bf(v.x, v.y), pk2bf(v.z, v.w));
  }
}

template<bool BF16W>
__global__ __launch_bounds__(256) void router_fused(
    const float* __restrict__ x,
    const float* __restrict__ W1f, const unsigned short* __restrict__ W1b,
    const float* __restrict__ b1,
    const float* __restrict__ W2f, const unsigned short* __restrict__ W2b,
    const float* __restrict__ b2, const float* __restrict__ logtemp,
    float* __restrict__ out)
{
  extern __shared__ char smem[];
  const int tid  = threadIdx.x;
  const int lane = tid & 31;
  const int wv   = tid >> 5;
  const int l16  = lane & 15;
  const int hsel = lane >> 4;                 // 0: k 0-7,16-23; 1: k 8-15,24-31 of each 32-k step
  const long r0  = (long)blockIdx.x * 128;

  // ---- stage full x tile: 128 rows x 1024 f32 -> bf16 LDS (read x once, NT) ----
  {
    const f32x4* xg = (const f32x4*)(x + r0 * 1024);
#pragma unroll 4
    for (int i = 0; i < 128; ++i) {
      f32x4 v = __builtin_nontemporal_load(&xg[(size_t)i * 256 + tid]);
      *(uint2*)(smem + XS_OFF + (size_t)i * (XS_STR*2) + (size_t)tid * 8) =
          make_uint2(pk2bf(v.x, v.y), pk2bf(v.z, v.w));
    }
  }

  const v8f zero = {0.f,0.f,0.f,0.f,0.f,0.f,0.f,0.f};
  v8f lacc[4]; // logits accumulators: 4 expert tiles of 16
#pragma unroll
  for (int t = 0; t < 4; ++t) lacc[t] = zero;

  const char* xs_w = smem + XS_OFF + (size_t)(wv*16 + l16) * (XS_STR*2);
  char* hs_w = smem + HS_OFF + wv * HS_WV;

  for (int hc = 0; hc < 16; ++hc) {
    const int h0 = hc * 64;
    v8f hacc[4];
#pragma unroll
    for (int t = 0; t < 4; ++t) hacc[t] = zero;

    for (int ks0 = 0; ks0 < 4; ++ks0) {
      __syncthreads();  // protect previous readers of W1/W2/HS LDS
      if constexpr (BF16W) {
#if HAS_TDM
        // Tensor Data Mover: strided global bf16 tile -> padded LDS rows.
        if (wv == 0) {
          // W1 block: 256 x 64 tile, row pad 8 bf16 (interval 128 DW, amount 4 DW)
          tdm_load_2d_bf16(W1_OFF, W1b + (size_t)h0 * 1024 + ks0 * 256,
                           1024, 1024, 1024, 256, 64, 6, 3);
          if (ks0 == 0) {
            // W2 block: 64 x 64 tile, row pad 8 bf16 (interval 32 DW, amount 4 DW)
            tdm_load_2d_bf16(W2_OFF, W2b + h0,
                             1024, 64, 1024, 64, 64, 4, 3);
          }
          __builtin_amdgcn_s_wait_tensorcnt(0);
        }
#else
        // plain bf16 copy: W1 64x256, 8 x b128 per thread
#pragma unroll
        for (int j = 0; j < 8; ++j) {
          int f4 = j * 256 + tid;
          int n  = f4 >> 5;
          int c16 = f4 & 31;
          uint4 v = ((const uint4*)W1b)[(size_t)(h0 + n) * 128 + ks0 * 32 + c16];
          *(uint4*)(smem + W1_OFF + n * (W1_STR*2) + c16 * 16) = v;
        }
        if (ks0 == 0) {
#pragma unroll
          for (int j = 0; j < 2; ++j) {
            int f4 = j * 256 + tid;
            int n  = f4 >> 3;
            int c16 = f4 & 7;
            uint4 v = ((const uint4*)W2b)[(size_t)n * 128 + (h0 >> 3) + c16];
            *(uint4*)(smem + W2_OFF + n * (W2_STR*2) + c16 * 16) = v;
          }
        }
#endif
      } else {
        // fp32 weights: stage + convert
#pragma unroll
        for (int j = 0; j < 16; ++j) {
          int f4 = j * 256 + tid;
          int n  = f4 >> 6;
          int c4 = f4 & 63;
          float4 v = *(const float4*)(W1f + (size_t)(h0 + n) * 1024 + ks0 * 256 + c4 * 4);
          *(uint2*)(smem + W1_OFF + n * (W1_STR*2) + c4 * 8) =
              make_uint2(pk2bf(v.x, v.y), pk2bf(v.z, v.w));
        }
        if (ks0 == 0) {
#pragma unroll
          for (int j = 0; j < 4; ++j) {
            int f4 = j * 256 + tid;
            int n  = f4 >> 4;
            int c4 = f4 & 15;
            float4 v = *(const float4*)(W2f + (size_t)n * 1024 + h0 + c4 * 4);
            *(uint2*)(smem + W2_OFF + n * (W2_STR*2) + c4 * 8) =
                make_uint2(pk2bf(v.x, v.y), pk2bf(v.z, v.w));
          }
        }
      }
      __syncthreads();

      // GEMM1: 8 K-steps of 32, 4 N-tiles each.
      // Preload A + all 4 B fragments into distinct registers so the scheduler
      // can keep several ds_load_b128 in flight across the WMMAs.
#pragma unroll
      for (int k8 = 0; k8 < 8; ++k8) {
        const int kk = ks0 * 256 + k8 * 32;
        const char* ap = xs_w + (size_t)(kk + hsel * 8) * 2;
        v16bf a = mk16(ldsv8(ap), ldsv8(ap + 32));
        v16bf b[4];
#pragma unroll
        for (int t = 0; t < 4; ++t) {
          const char* bp = smem + W1_OFF + (16*t + l16) * (W1_STR*2)
                         + (k8*32 + hsel*8) * 2;
          b[t] = mk16(ldsv8(bp), ldsv8(bp + 32));
        }
#pragma unroll
        for (int t = 0; t < 4; ++t) {
          hacc[t] = __builtin_amdgcn_wmma_f32_16x16x32_bf16(
              false, a, false, b[t], (short)0, hacc[t], false, false);
        }
      }
    }

    // bias + exact GELU, write 16x64 bf16 tile (lane-transpose via LDS)
#pragma unroll
    for (int t = 0; t < 4; ++t) {
      float bias = b1[h0 + 16*t + l16];
#pragma unroll
      for (int r = 0; r < 8; ++r) {
        float g = gelu_exact(hacc[t][r] + bias);
        int m = r + (hsel << 3);
        *(unsigned short*)(hs_w + m * HS_STR + (16*t + l16) * 2) = f2bf(g);
      }
    }
    __syncthreads();

    // GEMM2: logits += h(16x64) @ W2chunk^T  -> 2 sub-K x 4 expert tiles
#pragma unroll
    for (int s = 0; s < 2; ++s) {
      const char* ap = hs_w + l16 * HS_STR + (s*32 + hsel*8) * 2;
      v16bf a = mk16(ldsv8(ap), ldsv8(ap + 32));
      v16bf b[4];
#pragma unroll
      for (int t = 0; t < 4; ++t) {
        const char* bp = smem + W2_OFF + (16*t + l16) * (W2_STR*2)
                       + (s*32 + hsel*8) * 2;
        b[t] = mk16(ldsv8(bp), ldsv8(bp + 32));
      }
#pragma unroll
      for (int t = 0; t < 4; ++t) {
        lacc[t] = __builtin_amdgcn_wmma_f32_16x16x32_bf16(
            false, a, false, b[t], (short)0, lacc[t], false, false);
      }
    }
  }

  // ---- epilogue: logits -> LDS (overlay x region, now dead) ----
  __syncthreads();
#pragma unroll
  for (int t = 0; t < 4; ++t)
#pragma unroll
    for (int r = 0; r < 8; ++r) {
      int m = wv*16 + r + (hsel << 3);
      *(float*)(smem + LG_OFF + m * (LG_STR*4) + (16*t + l16) * 4) = lacc[t][r];
    }
  __syncthreads();

  // one thread per row: (v+b2)/temp, top-2, 2-way softmax
  if (tid < 128) {
    float tmp = expf(logtemp[0]);
    tmp = fminf(fmaxf(tmp, 1e-3f), 100.0f);
    float inv = 1.0f / tmp;
    const float* row = (const float*)(smem + LG_OFF + tid * (LG_STR*4));
    float m1 = -3.4e38f, m2 = -3.4e38f; int i1 = 0, i2 = 0;
    for (int e = 0; e < 64; ++e) {
      float v = (row[e] + b2[e]) * inv;
      if (v > m1)      { m2 = m1; i2 = i1; m1 = v; i1 = e; }
      else if (v > m2) { m2 = v;  i2 = e; }
    }
    float d  = expf(m2 - m1);
    float r1 = 1.0f / (1.0f + d);
    *(float4*)(smem + RI_OFF + tid * 16) =
        make_float4(r1, d * r1, __int_as_float(i1), __int_as_float(i2));
  }
  __syncthreads();

  // coalesced b128 NT output: zeros except the two selected experts per row
  f32x4* og = (f32x4*)(out + r0 * 64);
#pragma unroll
  for (int j = 0; j < 8; ++j) {
    int f4   = j * 256 + tid;
    int rrow = f4 >> 4;
    int c4   = f4 & 15;
    float4 ri = *(const float4*)(smem + RI_OFF + rrow * 16);
    int i1 = __float_as_int(ri.z), i2 = __float_as_int(ri.w);
    int e0 = c4 * 4;
    f32x4 o;
    o.x = (e0+0==i1) ? ri.x : ((e0+0==i2) ? ri.y : 0.0f);
    o.y = (e0+1==i1) ? ri.x : ((e0+1==i2) ? ri.y : 0.0f);
    o.z = (e0+2==i1) ? ri.x : ((e0+2==i2) ? ri.y : 0.0f);
    o.w = (e0+3==i1) ? ri.x : ((e0+3==i2) ? ri.y : 0.0f);
    __builtin_nontemporal_store(o, &og[f4]);
  }
}

extern "C" void kernel_launch(void* const* d_in, const int* in_sizes, int n_in,
                              void* d_out, int out_size, void* d_ws, size_t ws_size,
                              hipStream_t stream) {
  (void)n_in; (void)out_size;
  const float* x  = (const float*)d_in[0];
  const float* W1 = (const float*)d_in[1];
  const float* b1 = (const float*)d_in[2];
  const float* W2 = (const float*)d_in[3];
  const float* b2 = (const float*)d_in[4];
  const float* lt = (const float*)d_in[5];
  float* out = (float*)d_out;
  const int N = in_sizes[0] / 1024;           // 65536
  dim3 grid(N / 128), block(256);

  const size_t W1_BYTES = (size_t)1024 * 1024 * 2;   // 2 MB bf16
  const size_t W2_BYTES = (size_t)64 * 1024 * 2;     // 128 KB bf16
  if (ws_size >= W1_BYTES + W2_BYTES) {
    unsigned short* w1b = (unsigned short*)d_ws;
    unsigned short* w2b = (unsigned short*)((char*)d_ws + W1_BYTES);
    const int n4_w1 = 1024 * 1024 / 4, n4_w2 = 64 * 1024 / 4;
    hipLaunchKernelGGL(cvt_w_bf16, dim3((n4_w1 + 255) / 256), dim3(256), 0, stream,
                       (const float4*)W1, (uint2*)w1b, n4_w1);
    hipLaunchKernelGGL(cvt_w_bf16, dim3((n4_w2 + 255) / 256), dim3(256), 0, stream,
                       (const float4*)W2, (uint2*)w2b, n4_w2);
    hipLaunchKernelGGL(router_fused<true>, grid, block, SMEM_BYTES, stream,
                       x, (const float*)nullptr, w1b, b1,
                       (const float*)nullptr, w2b, b2, lt, out);
  } else {
    hipLaunchKernelGGL(router_fused<false>, grid, block, SMEM_BYTES, stream,
                       x, W1, (const unsigned short*)nullptr, b1,
                       W2, (const unsigned short*)nullptr, b2, lt, out);
  }
}
